// HabituatedEncoderLayer_28492813042055
// MI455X (gfx1250) — compile-verified, matrix operand order checked by reading, not run on hardware
//
#include <hip/hip_runtime.h>

// ---------------------------------------------------------------------------
// Types for CDNA5 WMMA (wave32)
// ---------------------------------------------------------------------------
typedef __attribute__((ext_vector_type(16))) __bf16 v16bf;
typedef __attribute__((ext_vector_type(8)))  __bf16 v8bf;
typedef __attribute__((ext_vector_type(8)))  float  v8f;

#define D_MODEL 1024
#define N_HEAD  16
#define HEAD_D  64
#define D_FF    4096
#define BATCH   2
#define SEQ_T   2048
#define M_ROWS  (BATCH * SEQ_T)   // 4096

__device__ __forceinline__ v8f wmma_bf16(v16bf a, v16bf b, v8f c) {
  // D = A(16x32 bf16) * B(32x16 bf16) + C(16x16 f32)
  return __builtin_amdgcn_wmma_f32_16x16x32_bf16(
      /*neg_a=*/false, a, /*neg_b=*/false, b,
      /*c_mod=*/(short)0, c, /*reuse_a=*/false, /*reuse_b=*/false);
}

// Load one 16x32 bf16 fragment (A layout; B matrices are stored N-major [N,K]
// row-major == column-of-B contiguous, so the same pattern applies).
// rowptr must already point at &Mat[row (= lane&15)][k0 + (lane>>4)*8].
__device__ __forceinline__ v16bf load_frag(const __bf16* rowptr) {
  v8bf lo = *(const v8bf*)(rowptr);       // K offsets {0..7}   (+half*8)
  v8bf hi = *(const v8bf*)(rowptr + 16);  // K offsets {16..23} (+half*8)
  v16bf f;
#pragma unroll
  for (int i = 0; i < 8; ++i) { f[i] = lo[i]; f[i + 8] = hi[i]; }
  return f;
}

__device__ __forceinline__ v8f v8f_zero() {
  v8f z = {0.f, 0.f, 0.f, 0.f, 0.f, 0.f, 0.f, 0.f};
  return z;
}

__device__ __forceinline__ float gelu_tanh(float x) {
  // jax.nn.gelu default (approximate=True)
  const float k0 = 0.7978845608028654f;   // sqrt(2/pi)
  const float k1 = 0.044715f;
  float x3 = x * x * x;
  return 0.5f * x * (1.0f + tanhf(k0 * (x + k1 * x3)));
}

// ---------------------------------------------------------------------------
// 16-lane butterfly exchange. gfx1250 wave32: v_permlane16_b32 permutes within
// rows of 16 lanes (exactly our reduction domain), pure VALU — avoids the
// ds_bpermute round-trips __shfl_xor lowers to.
// ---------------------------------------------------------------------------
#if __has_builtin(__builtin_amdgcn_permlane16)
template <int M>
__device__ __forceinline__ float permxor16(float x) {
  constexpr unsigned lo =
      ((0 ^ M) & 15) | (((1 ^ M) & 15) << 4) | (((2 ^ M) & 15) << 8) |
      (((3 ^ M) & 15) << 12) | (((4 ^ M) & 15) << 16) | (((5 ^ M) & 15) << 20) |
      (((6 ^ M) & 15) << 24) | (((7 ^ M) & 15) << 28);
  constexpr unsigned hi =
      ((8 ^ M) & 15) | (((9 ^ M) & 15) << 4) | (((10 ^ M) & 15) << 8) |
      (((11 ^ M) & 15) << 12) | (((12 ^ M) & 15) << 16) |
      (((13 ^ M) & 15) << 20) | (((14 ^ M) & 15) << 24) |
      (((15 ^ M) & 15) << 28);
  int xi = __float_as_int(x);
  int r = __builtin_amdgcn_permlane16(xi, xi, (int)lo, (int)hi, false, false);
  return __int_as_float(r);
}
#else
template <int M>
__device__ __forceinline__ float permxor16(float x) {
  return __shfl_xor(x, M, 32);   // masks <16 stay within each 16-lane half
}
#endif

__device__ __forceinline__ float red_max16(float x) {
  x = fmaxf(x, permxor16<1>(x));
  x = fmaxf(x, permxor16<2>(x));
  x = fmaxf(x, permxor16<4>(x));
  x = fmaxf(x, permxor16<8>(x));
  return x;
}
__device__ __forceinline__ float red_sum16(float x) {
  x += permxor16<1>(x);
  x += permxor16<2>(x);
  x += permxor16<4>(x);
  x += permxor16<8>(x);
  return x;
}

// ---------------------------------------------------------------------------
// f32 -> bf16 cast
// ---------------------------------------------------------------------------
__global__ __launch_bounds__(256) void cast_f32_bf16(const float* __restrict__ in,
                                                     __bf16* __restrict__ out,
                                                     size_t n) {
  size_t i = (size_t)blockIdx.x * blockDim.x + threadIdx.x;
  size_t stride = (size_t)gridDim.x * blockDim.x;
  for (; i < n; i += stride) out[i] = (__bf16)in[i];
}

// ---------------------------------------------------------------------------
// mask = causal - exp(log_tau) * mean_h(ema)   (T x T), memory-bound (~270MB
// read @ 23.3 TB/s is the hard floor of this layer)
// ---------------------------------------------------------------------------
__global__ __launch_bounds__(256) void mask_kernel(const float* __restrict__ causal,
                                                   const float* __restrict__ ema,
                                                   const float* __restrict__ log_tau,
                                                   float* __restrict__ mask,
                                                   size_t TT) {
  float tau = __expf(log_tau[0]);
  size_t i = (size_t)blockIdx.x * blockDim.x + threadIdx.x;
  size_t stride = (size_t)gridDim.x * blockDim.x;
  for (; i < TT; i += stride) {
    float s = 0.f;
#pragma unroll
    for (int h = 0; h < N_HEAD; ++h) s += ema[(size_t)h * TT + i];
    mask[i] = causal[i] - tau * (s * (1.0f / N_HEAD));
  }
}

// ---------------------------------------------------------------------------
// WMMA GEMM:  C[M,N] = A[M,K] @ W[N,K]^T + bias[N]   (optional GELU)
// A, W bf16 row-major; C is f32 or bf16.
// Block: 256 threads = 8 waves; block tile 128x128; wave tile 32(M) x 64(N);
// K-step 32. All operands (<=32MB bf16) are L2-resident on MI455X (192MB L2)
// so direct global fragment loads are fine; A frags reused across 4 N-tiles.
// ---------------------------------------------------------------------------
template <bool GELU, bool OUT_BF16>
__global__ __launch_bounds__(256) void gemm_bf16_kernel(const __bf16* __restrict__ A,
                                                        const __bf16* __restrict__ W,
                                                        const float* __restrict__ bias,
                                                        float* __restrict__ Cf,
                                                        __bf16* __restrict__ Cb,
                                                        int M, int N, int K) {
  const int wave = threadIdx.x >> 5;
  const int lane = threadIdx.x & 31;
  const int lm = lane & 15;
  const int half = lane >> 4;
  const int wm = wave & 3;        // 4 waves along M
  const int wn = wave >> 2;       // 2 waves along N
  const int m0 = blockIdx.y * 128 + wm * 32;
  const int n0 = blockIdx.x * 128 + wn * 64;

  v8f acc[2][4];
#pragma unroll
  for (int i = 0; i < 2; ++i)
#pragma unroll
    for (int j = 0; j < 4; ++j) acc[i][j] = v8f_zero();

  const __bf16* aRow0 = A + (size_t)(m0 + lm) * K + half * 8;
  const __bf16* aRow1 = A + (size_t)(m0 + 16 + lm) * K + half * 8;

  for (int k0 = 0; k0 < K; k0 += 32) {
    v16bf a0 = load_frag(aRow0 + k0);
    v16bf a1 = load_frag(aRow1 + k0);
    v16bf bf[4];
#pragma unroll
    for (int j = 0; j < 4; ++j) {
      const __bf16* wp = W + (size_t)(n0 + j * 16 + lm) * K + k0 + half * 8;
      bf[j] = load_frag(wp);
      if (k0 + 32 < K) __builtin_prefetch(wp + 32, 0, 2);   // global_prefetch_b8
    }
#pragma unroll
    for (int j = 0; j < 4; ++j) {
      acc[0][j] = wmma_bf16(a0, bf[j], acc[0][j]);
      acc[1][j] = wmma_bf16(a1, bf[j], acc[1][j]);
    }
  }

  // Epilogue. C/D layout: lane holds N = n0+j*16+lm; slot r -> M = r + half*8.
#pragma unroll
  for (int j = 0; j < 4; ++j) {
    const int n = n0 + j * 16 + lm;
    const float bz = bias ? bias[n] : 0.f;
#pragma unroll
    for (int mi = 0; mi < 2; ++mi) {
#pragma unroll
      for (int r = 0; r < 8; ++r) {
        const int row = m0 + mi * 16 + r + half * 8;
        float v = acc[mi][j][r] + bz;
        if (GELU) v = gelu_tanh(v);
        if (OUT_BF16) Cb[(size_t)row * N + n] = (__bf16)v;
        else          Cf[(size_t)row * N + n] = v;
      }
    }
  }
}

// ---------------------------------------------------------------------------
// Flash attention (causal + additive mask), bf16 QK^T / PV via WMMA,
// online softmax in f32. Q/K/V stored as [B*T, D] bf16, head h = cols h*64..+63.
// Grid: (B*H, T/128); block = 256 threads = 8 waves; each wave owns 16 q-rows.
// Pipeline: next K/V tile's global loads are issued before the current tile's
// WMMAs, LDS commit happens after compute, one __syncthreads per iteration.
// V is staged transposed so the PV B-fragment is a contiguous ds read; the
// per-wave P buffer does the C->A layout swap (in-order DS within a wave, no
// barrier needed).
// ---------------------------------------------------------------------------
__global__ __launch_bounds__(256) void flash_attn_kernel(const __bf16* __restrict__ Q,
                                                         const __bf16* __restrict__ Km,
                                                         const __bf16* __restrict__ Vm,
                                                         const float* __restrict__ mask,
                                                         __bf16* __restrict__ ctx) {
  __shared__ __bf16 ldsK[2][32 * 64];     // [buf][key][feat]
  __shared__ __bf16 ldsVT[2][64 * 32];    // [buf][feat][key]
  __shared__ __bf16 ldsP[8][16 * 32];     // per-wave P tile [qrow][key]

  const int bh = blockIdx.x;
  const int b = bh / N_HEAD;
  const int h = bh % N_HEAD;
  const int q0b = blockIdx.y * 128;
  const int wave = threadIdx.x >> 5;
  const int lane = threadIdx.x & 31;
  const int lm = lane & 15;
  const int half = lane >> 4;
  const size_t baseRow = (size_t)b * SEQ_T;
  const int hoff = h * HEAD_D;
  const int qrow0 = q0b + wave * 16;

  // cooperative loader lane mapping (256 threads, 16B each)
  const int ldKey = threadIdx.x >> 3;        // 0..31
  const int ldFg  = (threadIdx.x & 7) * 8;   // feature group

  // issue phase: global loads only (overlaps with compute of current tile)
  auto tile_issue = [&](int s0, v8bf& kd, v8bf& vd) {
    const size_t gr = (baseRow + s0 + ldKey) * D_MODEL + hoff + ldFg;
    kd = *(const v8bf*)(Km + gr);
    vd = *(const v8bf*)(Vm + gr);
  };
  // commit phase: LDS stores (K row-major, V transposed)
  auto tile_commit = [&](int buf, v8bf kd, v8bf vd) {
    *(v8bf*)&ldsK[buf][ldKey * 64 + ldFg] = kd;
#pragma unroll
    for (int i = 0; i < 8; ++i) ldsVT[buf][(ldFg + i) * 32 + ldKey] = vd[i];
  };

  // Q fragments for this wave's 16x64 tile (loaded once).
  const __bf16* qptr = Q + (baseRow + qrow0 + lm) * D_MODEL + hoff + half * 8;
  v16bf qa0 = load_frag(qptr);        // features 0..31
  v16bf qa1 = load_frag(qptr + 32);   // features 32..63

  v8f o[4];
#pragma unroll
  for (int j = 0; j < 4; ++j) o[j] = v8f_zero();
  float mrow[8], lrow[8];
#pragma unroll
  for (int r = 0; r < 8; ++r) { mrow[r] = -3.0e38f; lrow[r] = 0.f; }

  const int sEnd = q0b + 128;   // causal: keys beyond block's max query skipped

  // prologue: stage tile 0
  {
    v8bf kd, vd;
    tile_issue(0, kd, vd);
    tile_commit(0, kd, vd);
  }
  __syncthreads();

  int cur = 0;
  for (int s0 = 0; s0 < sEnd; s0 += 32, cur ^= 1) {
    const bool haveNext = (s0 + 32) < sEnd;
    v8bf nkd, nvd;
    if (haveNext) tile_issue(s0 + 32, nkd, nvd);   // loads in flight during compute

    // --- scores: two 16x16 tiles (keys s0..s0+15, s0+16..s0+31) ---
    v8f sAcc[2];
#pragma unroll
    for (int jj = 0; jj < 2; ++jj) {
      const __bf16* kb = &ldsK[cur][(jj * 16 + lm) * 64 + half * 8];
      v8f c = v8f_zero();
      c = wmma_bf16(qa0, load_frag(kb), c);        // features 0..31
      c = wmma_bf16(qa1, load_frag(kb + 32), c);   // features 32..63
      sAcc[jj] = c;
    }

    // --- online softmax (row = slot r + half*8; cols across 16 lanes) ---
#pragma unroll
    for (int r = 0; r < 8; ++r) {
      const int qr = qrow0 + r + half * 8;
      float s0v = sAcc[0][r] * 0.125f + mask[(size_t)qr * SEQ_T + s0 + lm];
      float s1v = sAcc[1][r] * 0.125f + mask[(size_t)qr * SEQ_T + s0 + 16 + lm];
      const float mx = red_max16(fmaxf(s0v, s1v));
      const float nm = fmaxf(mrow[r], mx);
      const float scale = __expf(mrow[r] - nm);
      const float p0 = __expf(s0v - nm);
      const float p1 = __expf(s1v - nm);
      lrow[r] = lrow[r] * scale + red_sum16(p0 + p1);
      mrow[r] = nm;
      sAcc[0][r] = p0;
      sAcc[1][r] = p1;
#pragma unroll
      for (int j = 0; j < 4; ++j) o[j][r] *= scale;
    }

    // --- P: C-layout -> A-layout via per-wave LDS buffer (in-order DS) ---
    __bf16* pw = ldsP[wave];
#pragma unroll
    for (int r = 0; r < 8; ++r) {
      const int row = r + half * 8;
      pw[row * 32 + lm]      = (__bf16)sAcc[0][r];
      pw[row * 32 + 16 + lm] = (__bf16)sAcc[1][r];
    }

    // --- PV: P(16x32) @ V(32x64) accumulated into o[0..3] ---
    v16bf pa = load_frag(&pw[lm * 32 + half * 8]);
#pragma unroll
    for (int j = 0; j < 4; ++j) {
      v16bf vb = load_frag(&ldsVT[cur][(j * 16 + lm) * 32 + half * 8]);
      o[j] = wmma_bf16(pa, vb, o[j]);
    }

    if (haveNext) tile_commit(cur ^ 1, nkd, nvd);
    __syncthreads();   // next buffer visible; current buffer free for i+1 commit
  }

  // --- normalize and store ctx (bf16, feeds O-projection GEMM) ---
  __bf16* cp = ctx + (baseRow + qrow0) * D_MODEL + hoff;
#pragma unroll
  for (int r = 0; r < 8; ++r) {
    const float inv = 1.0f / lrow[r];
    const int row = r + half * 8;
#pragma unroll
    for (int j = 0; j < 4; ++j)
      cp[(size_t)row * D_MODEL + j * 16 + lm] = (__bf16)(o[j][r] * inv);
  }
}

// ---------------------------------------------------------------------------
// out = LayerNorm(X + Y) * g + beta ; optionally also emit bf16 copy.
// One block (256 threads) per row of 1024.
// ---------------------------------------------------------------------------
template <bool WRITE_BF16>
__global__ __launch_bounds__(256) void residual_ln_kernel(const float* __restrict__ X,
                                                          const float* __restrict__ Y,
                                                          const float* __restrict__ g,
                                                          const float* __restrict__ be,
                                                          float* __restrict__ Of,
                                                          __bf16* __restrict__ Ob) {
  const int tid = threadIdx.x;
  const size_t off = (size_t)blockIdx.x * D_MODEL;
  float t[4];
  float s = 0.f, sq = 0.f;
#pragma unroll
  for (int i = 0; i < 4; ++i) {
    const int c = tid + i * 256;
    const float v = X[off + c] + Y[off + c];
    t[i] = v; s += v; sq += v * v;
  }
#pragma unroll
  for (int m = 1; m < 32; m <<= 1) {
    s += __shfl_xor(s, m, 32);
    sq += __shfl_xor(sq, m, 32);
  }
  __shared__ float rs[8], rq[8];
  if ((tid & 31) == 0) { rs[tid >> 5] = s; rq[tid >> 5] = sq; }
  __syncthreads();
  float ts = 0.f, tq = 0.f;
#pragma unroll
  for (int i = 0; i < 8; ++i) { ts += rs[i]; tq += rq[i]; }
  const float mu = ts * (1.0f / D_MODEL);
  const float var = tq * (1.0f / D_MODEL) - mu * mu;
  const float rstd = rsqrtf(var + 1e-5f);
#pragma unroll
  for (int i = 0; i < 4; ++i) {
    const int c = tid + i * 256;
    const float o = (t[i] - mu) * rstd * g[c] + be[c];
    Of[off + c] = o;
    if (WRITE_BF16) Ob[off + c] = (__bf16)o;
  }
}

// ---------------------------------------------------------------------------
// Launcher
// ---------------------------------------------------------------------------
extern "C" void kernel_launch(void* const* d_in, const int* in_sizes, int n_in,
                              void* d_out, int out_size, void* d_ws, size_t ws_size,
                              hipStream_t stream) {
  (void)in_sizes; (void)n_in; (void)out_size; (void)ws_size;
  const float* x      = (const float*)d_in[0];
  const float* causal = (const float*)d_in[1];
  const float* ema    = (const float*)d_in[2];
  const float* ltau   = (const float*)d_in[3];
  const float* Wq = (const float*)d_in[4];  const float* bq = (const float*)d_in[5];
  const float* Wk = (const float*)d_in[6];  const float* bk = (const float*)d_in[7];
  const float* Wv = (const float*)d_in[8];  const float* bv = (const float*)d_in[9];
  const float* Wo = (const float*)d_in[10]; const float* bo = (const float*)d_in[11];
  const float* W1 = (const float*)d_in[12]; const float* b1 = (const float*)d_in[13];
  const float* W2 = (const float*)d_in[14]; const float* b2 = (const float*)d_in[15];
  const float* g1 = (const float*)d_in[16]; const float* be1 = (const float*)d_in[17];
  const float* g2 = (const float*)d_in[18]; const float* be2 = (const float*)d_in[19];
  float* out = (float*)d_out;

  // workspace carve-out (256B aligned)
  char* w = (char*)d_ws;
  auto alloc = [&](size_t bytes) -> char* {
    char* p = w;
    w += (bytes + 255) & ~(size_t)255;
    return p;
  };
  const size_t MD  = (size_t)M_ROWS * D_MODEL;   // 4M elems
  const size_t DD  = (size_t)D_MODEL * D_MODEL;  // 1M
  const size_t FD  = (size_t)D_FF * D_MODEL;     // 4M
  const size_t MF  = (size_t)M_ROWS * D_FF;      // 16M
  const size_t TT  = (size_t)SEQ_T * SEQ_T;      // 4M

  __bf16* xb   = (__bf16*)alloc(MD * 2);
  __bf16* wqb  = (__bf16*)alloc(DD * 2);
  __bf16* wkb  = (__bf16*)alloc(DD * 2);
  __bf16* wvb  = (__bf16*)alloc(DD * 2);
  __bf16* wob  = (__bf16*)alloc(DD * 2);
  __bf16* w1b  = (__bf16*)alloc(FD * 2);
  __bf16* w2b  = (__bf16*)alloc(FD * 2);
  __bf16* qb   = (__bf16*)alloc(MD * 2);
  __bf16* kb   = (__bf16*)alloc(MD * 2);
  __bf16* vb   = (__bf16*)alloc(MD * 2);
  float*  msk  = (float*)alloc(TT * 4);
  __bf16* ctx  = (__bf16*)alloc(MD * 2);
  float*  attn = (float*)alloc(MD * 4);
  float*  hx   = (float*)alloc(MD * 4);
  __bf16* hb   = (__bf16*)alloc(MD * 2);
  __bf16* ff1  = (__bf16*)alloc(MF * 2);
  float*  ff2  = (float*)alloc(MD * 4);

  // 1) bf16 casts
  cast_f32_bf16<<<2048, 256, 0, stream>>>(x,  xb,  MD);
  cast_f32_bf16<<<1024, 256, 0, stream>>>(Wq, wqb, DD);
  cast_f32_bf16<<<1024, 256, 0, stream>>>(Wk, wkb, DD);
  cast_f32_bf16<<<1024, 256, 0, stream>>>(Wv, wvb, DD);
  cast_f32_bf16<<<1024, 256, 0, stream>>>(Wo, wob, DD);
  cast_f32_bf16<<<2048, 256, 0, stream>>>(W1, w1b, FD);
  cast_f32_bf16<<<2048, 256, 0, stream>>>(W2, w2b, FD);

  // 2) attn mask (causal + habituation bias)
  mask_kernel<<<8192, 256, 0, stream>>>(causal, ema, ltau, msk, TT);

  // 3) QKV projections (bf16 out)
  dim3 gQKV(D_MODEL / 128, M_ROWS / 128);
  gemm_bf16_kernel<false, true><<<gQKV, 256, 0, stream>>>(xb, wqb, bq, nullptr, qb,
                                                          M_ROWS, D_MODEL, D_MODEL);
  gemm_bf16_kernel<false, true><<<gQKV, 256, 0, stream>>>(xb, wkb, bk, nullptr, kb,
                                                          M_ROWS, D_MODEL, D_MODEL);
  gemm_bf16_kernel<false, true><<<gQKV, 256, 0, stream>>>(xb, wvb, bv, nullptr, vb,
                                                          M_ROWS, D_MODEL, D_MODEL);

  // 4) flash attention
  dim3 gFA(BATCH * N_HEAD, SEQ_T / 128);
  flash_attn_kernel<<<gFA, 256, 0, stream>>>(qb, kb, vb, msk, ctx);

  // 5) output projection (f32 out)
  gemm_bf16_kernel<false, false><<<gQKV, 256, 0, stream>>>(ctx, wob, bo, attn, nullptr,
                                                           M_ROWS, D_MODEL, D_MODEL);

  // 6) h = LN(x + attn_out); keep f32 + bf16
  residual_ln_kernel<true><<<M_ROWS, 256, 0, stream>>>(x, attn, g1, be1, hx, hb);

  // 7) FFN
  dim3 gF1(D_FF / 128, M_ROWS / 128);
  gemm_bf16_kernel<true, true><<<gF1, 256, 0, stream>>>(hb, w1b, b1, nullptr, ff1,
                                                        M_ROWS, D_FF, D_MODEL);
  gemm_bf16_kernel<false, false><<<gQKV, 256, 0, stream>>>(ff1, w2b, b2, ff2, nullptr,
                                                           M_ROWS, D_MODEL, D_FF);

  // 8) out = LN(h + ff)
  residual_ln_kernel<false><<<M_ROWS, 256, 0, stream>>>(hx, ff2, g2, be2, out, nullptr);
}
